// VocabGraphConvolution_10831907520811
// MI455X (gfx1250) — matrix operand order repaired and checked by reading, not compiled, not run on hardware
//
#include <hip/hip_runtime.h>

// ---------------------------------------------------------------------------
// VocabGraphConvolution for MI455X (gfx1250, wave32).
//   V=30522, A=2, H=128, O=16, M=B*D=2048, NNZ=250000.
//
// Roofline: main GEMM x[2048,V] @ h_sum[V,128] reads 250 MB of x once
// (~10.7 us at 23.3 TB/s). f16 WMMA (16x16x32) keeps compute under that
// floor. To reach the bandwidth floor we need ~15 MB of loads in flight
// (23.3 TB/s x ~700 ns), so the GEMM is split 64 M-blocks x 8 K-chunks
// = 512 workgroups, each double-buffered — partial tiles accumulate into a
// fused_h buffer with native f32 atomics. Each wave keeps TWO 16x16 M-tiles
// per B fragment (M-block = 32 rows) to halve packed-B L2 traffic (~0.5 GB).
// x streams with TH=NT so the 7.8 MB packed B stays L2-resident.
//
// Pipeline:
//   1. zero_buf x2 : clear h_sum[V,H] f32 and fused[2048,128] f32
//   2. spmm        : COO scatter-add, one wave/nonzero, native fp32 atomics
//   3. pack_b      : h_sum f32 -> f16 in exact WMMA B-fragment layout,
//                    zero-padded to K=30592
//   4. gemm        : v_wmma_f32_16x16x32_f16, dual accumulators, K-split,
//                    cross-iteration B prefetch, atomic f32 epilogue
//   5. proj        : out = fused @ fc_W + fc_b  -> out[2048,16]
//
// Schedule notes: B fragments for stage i+1 are loaded into a second register
// set DURING stage i's WMMAs (cross-iteration dataflow forces the compiler to
// keep a full stage of latency slack; it cannot sink the loads to their
// uses). Stage-movement VALU (cvt/min/addr) is hoisted before the WMMA block.
// ---------------------------------------------------------------------------

#define V_N     30522
#define A_N     2
#define H_N     128
#define O_N     16
#define M_N     2048
#define NNZ_N   250000
#define KC      128                 // K per LDS stage
#define NSTAGES 239                 // ceil(V/128)
#define VPAD    (NSTAGES * KC)      // 30592
#define NKTILES (VPAD / 32)         // 956 K-tiles of 32
#define GCMAX   30512               // largest 8-aligned c with c+8 <= V
#define MBLK    32                  // M rows per block (2 tiles per wave)
#define KSPLIT  8                   // K-chunks (blocks) per M-block
#define SCHUNK  30                  // ceil(NSTAGES / KSPLIT)

typedef __attribute__((ext_vector_type(16))) _Float16 v16h;
typedef __attribute__((ext_vector_type(8)))  float    v8f;
typedef __attribute__((ext_vector_type(4)))  unsigned v4u;
typedef __attribute__((ext_vector_type(4)))  float    v4f;

union ABu { v4u q[2]; v16h v; };    // 32B WMMA fragment reinterpret

__device__ __forceinline__ unsigned pk2h(float a, float b) {
  return __builtin_bit_cast(unsigned, __builtin_amdgcn_cvt_pkrtz(a, b));
}

// ---------------------------------------------------------------- 1: zero ws
__global__ void zero_buf(float* __restrict__ p, unsigned n) {
  unsigned i = blockIdx.x * blockDim.x + threadIdx.x;
  const unsigned step = gridDim.x * blockDim.x;
  for (; i < n; i += step) p[i] = 0.f;
}

// ---------------------------------------------------------------- 2: SpMM
// One wave per nonzero edge; lane l owns h = 4*l..4*l+3 (H=128 = 32 lanes x 4).
__global__ __launch_bounds__(256) void spmm(
    const int*   __restrict__ rows, const int* __restrict__ cols,
    const float* __restrict__ vals, const float* __restrict__ W,
    float* __restrict__ hsum) {
  const int wid  = (int)((blockIdx.x * blockDim.x + threadIdx.x) >> 5);
  const int lane = threadIdx.x & 31;
  if (wid >= A_N * NNZ_N) return;
  const int a = wid / NNZ_N;                 // rows/cols/vals are [A,NNZ] flat
  const int r = rows[wid];
  const int c = cols[wid];
  const float v = vals[wid];
  const v4f wv = *(const v4f*)(W + ((size_t)a * V_N + c) * H_N + lane * 4);
  float* dst = hsum + (size_t)r * H_N + lane * 4;
  unsafeAtomicAdd(dst + 0, v * wv.x);
  unsafeAtomicAdd(dst + 1, v * wv.y);
  unsafeAtomicAdd(dst + 2, v * wv.z);
  unsafeAtomicAdd(dst + 3, v * wv.w);
}

// ------------------------------------------------------- 3: pack B fragments
// B fragment layout chosen here (contiguous k, matching A) for v_wmma_*_16x16x32 (16-bit B, 32x16):
//   lane l: column n = l&15, halfB = l>>4; element e (0..15): K = halfB*16+e.
// Tile (kc, nt) stored lane-major: 32 lanes x 16 f16 = 1KB, so the GEMM's
// per-lane fragment load is two coalesced b128 reads. K in [V, VPAD) -> 0.
__global__ __launch_bounds__(256) void pack_b(
    const float* __restrict__ hsum, unsigned short* __restrict__ Bp) {
  const int kc    = blockIdx.x;         // 0..NKTILES-1 (K chunk of 32)
  const int nt    = threadIdx.x >> 5;   // 0..7 (N tile of 16)
  const int lane  = threadIdx.x & 31;
  const int n     = nt * 16 + (lane & 15);
  const int halfB = lane >> 4;
  unsigned pk[8];
#pragma unroll
  for (int p = 0; p < 8; ++p) {
    const int k0 = kc * 32 + halfB * 16 + 2 * p;
    const float f0 = (k0     < V_N) ? hsum[(size_t)k0 * H_N + n]       : 0.f;
    const float f1 = (k0 + 1 < V_N) ? hsum[(size_t)(k0 + 1) * H_N + n] : 0.f;
    pk[p] = pk2h(f0, f1);
  }
  v4u* dst = (v4u*)(Bp + ((size_t)kc * 8 + nt) * 512 + lane * 16);
  v4u q0 = {pk[0], pk[1], pk[2], pk[3]};
  v4u q1 = {pk[4], pk[5], pk[6], pk[7]};
  dst[0] = q0;
  dst[1] = q1;
}

// ------------------------------------------------------------------ 4: GEMM
// Block (mb, kc): 32 M-rows x N=128 x K-chunk of <=30 stages. 8 waves, wave w
// owns N tile w with two accumulators (M 0-15 / 16-31) sharing each B frag.
__global__ __launch_bounds__(256) void gemm(
    const float* __restrict__ x, const unsigned short* __restrict__ Bp,
    float* __restrict__ fused) {
  __shared__ unsigned short ldsA[2][MBLK * KC];  // A stages, f16, 2 x 8KB

  const int t    = threadIdx.x;
  const int wave = t >> 5;
  const int lane = t & 31;
  const int m0   = blockIdx.x * MBLK;
  const int mrow = lane & 15;
  const int hlf  = lane >> 4;

  const int sBeg = blockIdx.y * SCHUNK;
  const int sEnd = min(NSTAGES, sBeg + SCHUNK);
  const int nIt  = sEnd - sBeg;

  // staging: thread t owns row (t>>3), 16 contiguous cols at (t&7)*16
  const int arow  = t >> 3;
  const int acol0 = (t & 7) * 16;
  const float* xrow = x + (size_t)(m0 + arow) * V_N;

  v4f f0, f1, f2, f3;  // in-flight stage data (registers)
  auto fetch = [&](int k0) {
    // branch-free: clamp to last aligned in-bounds chunk; values beyond V are
    // don't-cares (B rows >= V are zero), they just must be mapped memory.
    const int g0 = min(k0 + acol0, GCMAX);
    const int g1 = min(k0 + acol0 + 8, GCMAX);
    f0 = __builtin_nontemporal_load((const v4f*)(xrow + g0));      // TH=NT
    f1 = __builtin_nontemporal_load((const v4f*)(xrow + g0 + 4));
    f2 = __builtin_nontemporal_load((const v4f*)(xrow + g1));
    f3 = __builtin_nontemporal_load((const v4f*)(xrow + g1 + 4));
  };
  auto commit = [&](int buf) {  // f32 -> packed f16, two b128 LDS stores
    v4u q0 = {pk2h(f0.x, f0.y), pk2h(f0.z, f0.w),
              pk2h(f1.x, f1.y), pk2h(f1.z, f1.w)};
    v4u q1 = {pk2h(f2.x, f2.y), pk2h(f2.z, f2.w),
              pk2h(f3.x, f3.y), pk2h(f3.z, f3.w)};
    v4u* p = (v4u*)&ldsA[buf][arow * KC + acol0];
    p[0] = q0;
    p[1] = q1;
  };
  auto loadB = [](ABu& x0, ABu& x1, ABu& x2, ABu& x3,
                  const unsigned short* base) {
    const v4u* p0 = (const v4u*)(base);
    const v4u* p1 = (const v4u*)(base + 4096);
    const v4u* p2 = (const v4u*)(base + 8192);
    const v4u* p3 = (const v4u*)(base + 12288);
    x0.q[0] = p0[0]; x0.q[1] = p0[1];
    x1.q[0] = p1[0]; x1.q[1] = p1[1];
    x2.q[0] = p2[0]; x2.q[1] = p2[1];
    x3.q[0] = p3[0]; x3.q[1] = p3[1];
  };

  fetch(sBeg * KC);
  commit(0);
  fetch((sBeg + 1) * KC);  // clamped: safe even if chunk has 1 stage
  __syncthreads();         // buf0 visible

  v8f acc0 = {0.f, 0.f, 0.f, 0.f, 0.f, 0.f, 0.f, 0.f};
  v8f acc1 = {0.f, 0.f, 0.f, 0.f, 0.f, 0.f, 0.f, 0.f};

  // per-wave B base: stage stride = 4 subtiles * 8 ntiles * 512 = 16384 shorts
  const unsigned short* bwl =
      Bp + (size_t)sBeg * 16384 + wave * 512 + lane * 16;

  ABu bc0, bc1, bc2, bc3;  // compute set (this stage)
  ABu bn0, bn1, bn2, bn3;  // prefetch set (next stage)
  loadB(bc0, bc1, bc2, bc3, bwl);

#pragma unroll 2
  for (int i = 0; i < nIt; ++i) {
    // next-stage B in flight across the whole stage (cross-iteration use
    // forces the loads to stay a full stage ahead of their consumers; the
    // final prefetch over-reads into the adjacent fused buffer: mapped,
    // never consumed)
    loadB(bn0, bn1, bn2, bn3, bwl + 16384);

    // stage movement (VALU + stores + NT loads, overlaps WMMAs)
    commit((i + 1) & 1);              // other buffer: prev barrier ended reads
    fetch((sBeg + i + 2) * KC);       // clamped over-fetch is harmless

    // 4 substeps x 2 WMMAs; A frags from LDS with immediate offsets
    const unsigned short* abase = &ldsA[i & 1][0];
    const unsigned short* la0 = abase + mrow * KC + hlf * 16;
    const unsigned short* la1 = la0 + 16 * KC;
    auto step = [&](const ABu& b, int koff) {
      ABu a0, a1;
      const v4u* pa0 = (const v4u*)(la0 + koff);
      a0.q[0] = pa0[0];
      a0.q[1] = pa0[1];  // +8 f16 elements = +16B
      const v4u* pa1 = (const v4u*)(la1 + koff);
      a1.q[0] = pa1[0];
      a1.q[1] = pa1[1];
      acc0 = __builtin_amdgcn_wmma_f32_16x16x32_f16(
          false, a0.v, false, b.v, (short)0, acc0, false, false);
      acc1 = __builtin_amdgcn_wmma_f32_16x16x32_f16(
          false, a1.v, false, b.v, (short)0, acc1, false, false);
    };
    step(bc0, 0);
    step(bc1, 32);
    step(bc2, 64);
    step(bc3, 96);

    bc0 = bn0;  // rotation vanishes under unroll-by-2 renaming
    bc1 = bn1;
    bc2 = bn2;
    bc3 = bn3;
    bwl += 16384;
    __syncthreads();
  }

  // atomic epilogue: add partial tiles into fused[2048,128]
  // (C layout: lane n = l&15; vgpr j -> m = hlf*8 + j)
  const int ncol = wave * 16 + (lane & 15);
#pragma unroll
  for (int j = 0; j < 8; ++j) {
    const int m = hlf * 8 + j;
    unsafeAtomicAdd(&fused[(size_t)(m0 + m) * H_N + ncol], acc0[j]);
    unsafeAtomicAdd(&fused[(size_t)(m0 + 16 + m) * H_N + ncol], acc1[j]);
  }
}

// ------------------------------------------------- 5: projection (tiny GEMM)
__global__ __launch_bounds__(256) void proj(
    const float* __restrict__ fused, const float* __restrict__ fcW,
    const float* __restrict__ fcb, float* __restrict__ out) {
  const int m = blockIdx.x * 16 + (threadIdx.x >> 4);
  const int o = threadIdx.x & 15;
  const float* fr = fused + (size_t)m * H_N;
  float sum = fcb[o];
#pragma unroll 8
  for (int h = 0; h < H_N; ++h) sum += fr[h] * fcW[h * O_N + o];
  out[(size_t)m * O_N + o] = sum;
}

// ---------------------------------------------------------------------------
extern "C" void kernel_launch(void* const* d_in, const int* in_sizes, int n_in,
                              void* d_out, int out_size, void* d_ws,
                              size_t ws_size, hipStream_t stream) {
  (void)in_sizes; (void)n_in; (void)out_size; (void)ws_size;
  const float* x    = (const float*)d_in[0];
  const int*   rows = (const int*)d_in[1];
  const int*   cols = (const int*)d_in[2];
  const float* vals = (const float*)d_in[3];
  const float* W    = (const float*)d_in[4];
  const float* fcW  = (const float*)d_in[5];
  const float* fcb  = (const float*)d_in[6];
  float* out = (float*)d_out;

  // workspace: [h_sum f32 : 15,627,264 B][B_pack f16 : 7,831,552 B]
  //            [fused f32 : 1,048,576 B]   total ~24.5 MB
  float* hsum = (float*)d_ws;
  unsigned short* Bp =
      (unsigned short*)((char*)d_ws + (size_t)V_N * H_N * sizeof(float));
  float* fused = (float*)((char*)Bp + (size_t)VPAD * H_N * sizeof(short));

  zero_buf<<<2048, 256, 0, stream>>>(hsum, V_N * H_N);
  zero_buf<<<256, 256, 0, stream>>>(fused, M_N * H_N);
  spmm<<<(A_N * NNZ_N + 7) / 8, 256, 0, stream>>>(rows, cols, vals, W, hsum);
  pack_b<<<NKTILES, 256, 0, stream>>>(hsum, Bp);
  gemm<<<dim3(M_N / MBLK, KSPLIT), 256, 0, stream>>>(x, Bp, fused);
  proj<<<M_N / 16, 256, 0, stream>>>(fused, fcW, fcb, out);
}
